// HSTU_10333691314795
// MI455X (gfx1250) — matrix operand order repaired and checked by reading, not compiled, not run on hardware
//
#include <hip/hip_runtime.h>
#include <hip/hip_bf16.h>

// Problem constants (match reference)
#define BB  2
#define LL  2048
#define HH  1024
#define NHH 16
#define HDD 64

typedef __attribute__((ext_vector_type(16))) __bf16 v16bf;
typedef __attribute__((ext_vector_type(8)))  __bf16 v8bf;
typedef __attribute__((ext_vector_type(8)))  float  v8f;

__device__ __forceinline__ unsigned short f2bf(float f) {
  unsigned int u = __float_as_uint(f);
  u += 0x7FFFu + ((u >> 16) & 1u);      // round-to-nearest-even
  return (unsigned short)(u >> 16);
}
__device__ __forceinline__ float bf2f(unsigned short s) {
  return __uint_as_float(((unsigned int)s) << 16);
}

// ---- WMMA operand loaders (contiguous k order, same in A and B) ----
// A tile 16xK-chunk(32), bf16 row-major with leading dim ld:
//   lanes 0-15 : row m=lane,    K = kc+0..15  (contiguous)
//   lanes16-31 : row m=lane-16, K = kc+16..31 (contiguous)
__device__ __forceinline__ v16bf load_a(const unsigned short* A, int row0, int ld,
                                        int kc, int lane) {
  const int l16 = lane & 15, hi = lane >> 4;
  const unsigned short* p = A + (size_t)(row0 + l16) * ld + kc + hi * 16;
  v8bf lo = *reinterpret_cast<const v8bf*>(p);
  v8bf hh = *reinterpret_cast<const v8bf*>(p + 8);
  v16bf r;
#pragma unroll
  for (int i = 0; i < 8; ++i) { r[i] = lo[i]; r[i + 8] = hh[i]; }
  return r;
}
// B tile 32x16, bf16 stored column-contiguous (Bt[n][k], leading dim ld):
//   lanes 0-15 : col n=lane,    K = kc+0..15  (contiguous)
//   lanes16-31 : col n=lane-16, K = kc+16..31 (contiguous)
__device__ __forceinline__ v16bf load_b(const unsigned short* Bt, int col0, int ld,
                                        int kc, int lane) {
  const int l16 = lane & 15, hi = lane >> 4;
  const unsigned short* p = Bt + (size_t)(col0 + l16) * ld + kc + hi * 16;
  return *reinterpret_cast<const v16bf*>(p);
}
__device__ __forceinline__ v8f wmma_bf16(v16bf a, v16bf b, v8f c) {
  return __builtin_amdgcn_wmma_f32_16x16x32_bf16(false, a, false, b, (short)0, c,
                                                 false, false);
}

// ---- Kernel 1: f32 [K][N] -> bf16 [N][K] (WMMA B-operand layout) ----
__global__ void __launch_bounds__(256)
k_wconv(const float* __restrict__ W, unsigned short* __restrict__ Wt, int Kd, int N) {
  int idx = blockIdx.x * blockDim.x + threadIdx.x;
  if (idx >= Kd * N) return;
  int k = idx / N, n = idx - k * N;
  Wt[(size_t)n * Kd + k] = f2bf(W[(size_t)k * N + n]);
}

// ---- Kernel 2: row layernorm f32[R][1024] -> bf16 ----
__global__ void __launch_bounds__(256)
k_ln(const float* __restrict__ x, const float* __restrict__ g,
     const float* __restrict__ be, unsigned short* __restrict__ out) {
  const int row = blockIdx.x, t = threadIdx.x;
  const float* xr = x + (size_t)row * HH;
  __shared__ float rs[256], rq[256];
  float s = 0.f, q = 0.f;
#pragma unroll
  for (int i = 0; i < 4; ++i) { float v = xr[t + 256 * i]; s += v; q += v * v; }
  rs[t] = s; rq[t] = q;
  __syncthreads();
  for (int off = 128; off > 0; off >>= 1) {
    if (t < off) { rs[t] += rs[t + off]; rq[t] += rq[t + off]; }
    __syncthreads();
  }
  const float mu   = rs[0] * (1.0f / HH);
  const float var  = rq[0] * (1.0f / HH) - mu * mu;
  const float rstd = rsqrtf(var + 1e-8f);
#pragma unroll
  for (int i = 0; i < 4; ++i) {
    int c = t + 256 * i;
    out[(size_t)row * HH + c] = f2bf((xr[c] - mu) * rstd * g[c] + be[c]);
  }
}

// ---- Kernel 3: bf16 WMMA GEMM, one wave -> 16x128 tile (N % 128 == 0) ----
// mode 0: out_bf = bf16(A@B + bias)
// mode 1: out_f  = residual + A@B + bias
__global__ void __launch_bounds__(32)
k_gemm(const unsigned short* __restrict__ A, const unsigned short* __restrict__ Bt,
       const float* __restrict__ bias, const float* __restrict__ residual,
       unsigned short* __restrict__ out_bf, float* __restrict__ out_f,
       int M, int N, int Kd, int mode) {
  const int lane = threadIdx.x;
  const int l16 = lane & 15, hi = lane >> 4;
  const int m0 = blockIdx.x * 16;
  const int n0 = blockIdx.y * 128;
  v8f acc[8] = {{}, {}, {}, {}, {}, {}, {}, {}};
  for (int kc = 0; kc < Kd; kc += 32) {
    // pull the tile four K-steps ahead toward the WGP (global_prefetch_b8);
    // guard is wave-uniform so EXEC stays all-ones around the WMMAs
    if (kc + 128 < Kd) {
      __builtin_prefetch(A + (size_t)(m0 + l16) * Kd + kc + 128, 0, 3);
#pragma unroll
      for (int c = 0; c < 8; ++c)
        __builtin_prefetch(Bt + (size_t)(n0 + c * 16 + l16) * Kd + kc + 128, 0, 3);
    }
    v16bf a = load_a(A, m0, Kd, kc, lane);     // A reused across 8 WMMAs
#pragma unroll
    for (int c = 0; c < 8; ++c)
      acc[c] = wmma_bf16(a, load_b(Bt, n0 + c * 16, Kd, kc, lane), acc[c]);
  }
  // C/D layout: VGPR r, lanes 0-15 -> M=r, lanes 16-31 -> M=r+8, N = lane&15
#pragma unroll
  for (int c = 0; c < 8; ++c) {
#pragma unroll
    for (int r = 0; r < 8; ++r) {
      int m = m0 + r + hi * 8;
      int col = n0 + c * 16 + l16;
      float v = acc[c][r] + bias[col];
      if (mode == 0) out_bf[(size_t)m * N + col] = f2bf(v);
      else           out_f[(size_t)m * N + col] =
                         residual[(size_t)m * N + col] + v;
    }
  }
}

// ---- Kernel 4: split proj -> RoPE(Q,K) row-major per head, V transposed ----
__global__ void __launch_bounds__(256)
k_split_rope(const unsigned short* __restrict__ proj,
             unsigned short* __restrict__ Qh, unsigned short* __restrict__ Kh,
             unsigned short* __restrict__ Vt) {
  int idx = blockIdx.x * blockDim.x + threadIdx.x;   // 2^22 total
  int d = idx & 63;
  int h = (idx >> 6) & 15;
  int l = (idx >> 10) & (LL - 1);
  int b = idx >> 21;
  size_t row = (size_t)b * LL + l;
  const unsigned short* pr = proj + row * 4096;
  const int hc = h * HDD;
  const int dp = (d < 32) ? d + 32 : d - 32;
  float qv = bf2f(pr[2048 + hc + d]);
  float qp = bf2f(pr[2048 + hc + dp]);
  float kv = bf2f(pr[3072 + hc + d]);
  float kp = bf2f(pr[3072 + hc + dp]);
  float vv = bf2f(pr[1024 + hc + d]);
  int   j   = d & 31;
  float inv = __powf(10000.0f, -(float)(2 * j) * (1.0f / 64.0f));
  float ang = (float)l * inv;
  float c = __cosf(ang), s = __sinf(ang);
  float rq = (d < 32) ? -qp : qp;
  float rk = (d < 32) ? -kp : kp;
  size_t hb = ((size_t)(b * NHH + h)) * LL;            // head base (rows)
  Qh[(hb + l) * HDD + d] = f2bf(qv * c + rq * s);
  Kh[(hb + l) * HDD + d] = f2bf(kv * c + rk * s);
  Vt[hb * HDD + (size_t)d * LL + l] = f2bf(vv);        // [HD][L] per head
}

// ---- Kernel 5: causal sigmoid attention + silu(U) gate + residual ----
// One wave per (batch, head, 16-query tile). Key blocks of 32.
__global__ void __launch_bounds__(32)
k_attn(const unsigned short* __restrict__ Qh, const unsigned short* __restrict__ Kh,
       const unsigned short* __restrict__ Vt, const unsigned short* __restrict__ proj,
       const float* __restrict__ x, float* __restrict__ x2) {
  const int lane = threadIdx.x;
  const int l16 = lane & 15, hi = lane >> 4;
  const int qt = blockIdx.x, h = blockIdx.y, b = blockIdx.z;
  const int q0 = qt * 16;
  const size_t hoff = ((size_t)(b * NHH + h)) * LL * HDD;
  const unsigned short* Q = Qh + hoff;
  const unsigned short* K = Kh + hoff;
  const unsigned short* V = Vt + hoff;                 // [HD][L]

  __shared__ __align__(32) unsigned short Pl[16 * 32]; // P tile bf16 [16][32]

  // Q tile resident in registers (A operand, K = 0..31 and 32..63)
  v16bf aq0 = load_a(Q, q0, HDD, 0, lane);
  v16bf aq1 = load_a(Q, q0, HDD, 32, lane);

  v8f o[4] = {{}, {}, {}, {}};
  const float scale = 0.125f;                          // HD^-0.5
  const int kbmax = (q0 + 15) / 32;                    // uniform across wave
  for (int kb = 0; kb <= kbmax; ++kb) {
    const int kbase = kb * 32;
    // S = Q K^T for 32 keys (two 16x16 tiles, K-dim = HD = 64)
    v8f s0 = {}, s1 = {};
    s0 = wmma_bf16(aq0, load_b(K, kbase,      HDD, 0,  lane), s0);
    s0 = wmma_bf16(aq1, load_b(K, kbase,      HDD, 32, lane), s0);
    s1 = wmma_bf16(aq0, load_b(K, kbase + 16, HDD, 0,  lane), s1);
    s1 = wmma_bf16(aq1, load_b(K, kbase + 16, HDD, 32, lane), s1);
    // causal mask + sigmoid, stage P into LDS as bf16 (C-layout -> row-major)
#pragma unroll
    for (int r = 0; r < 8; ++r) {
      const int m = r + hi * 8;
      const int qrow = q0 + m;
      const int key0 = kbase + l16;
      const int key1 = kbase + 16 + l16;
      float p0 = (key0 <= qrow) ? 1.0f / (1.0f + __expf(-s0[r] * scale)) : 0.0f;
      float p1 = (key1 <= qrow) ? 1.0f / (1.0f + __expf(-s1[r] * scale)) : 0.0f;
      Pl[m * 32 + l16]      = f2bf(p0);
      Pl[m * 32 + 16 + l16] = f2bf(p1);
    }
    __builtin_amdgcn_wave_barrier();
    asm volatile("s_wait_dscnt 0" ::: "memory");       // in-wave LDS RAW fence
    // Reload P in A-operand layout; O += P @ V  (K-dim = 32 keys)
    v16bf pa = load_a(Pl, 0, 32, 0, lane);
#pragma unroll
    for (int t = 0; t < 4; ++t)
      o[t] = wmma_bf16(pa, load_b(V, t * 16, LL, kbase, lane), o[t]);
    __builtin_amdgcn_wave_barrier();
  }
  // epilogue: x2 = x + silu(U) * attn_out
#pragma unroll
  for (int t = 0; t < 4; ++t) {
#pragma unroll
    for (int r = 0; r < 8; ++r) {
      const int m = r + hi * 8;
      const size_t grow = (size_t)b * LL + (q0 + m);
      const int col = h * HDD + t * 16 + l16;          // < 1024
      float u = bf2f(proj[grow * 4096 + col]);         // U = proj[:, 0:1024]
      float gated = (u / (1.0f + __expf(-u))) * o[t][r];
      x2[grow * HH + col] = x[grow * HH + col] + gated;
    }
  }
}

extern "C" void kernel_launch(void* const* d_in, const int* in_sizes, int n_in,
                              void* d_out, int out_size, void* d_ws, size_t ws_size,
                              hipStream_t stream) {
  const float* x      = (const float*)d_in[0];
  // d_in[1] = attn_mask (causal tril) — computed analytically, not read
  const float* W_proj = (const float*)d_in[2];
  const float* b_proj = (const float*)d_in[3];
  const float* W_out  = (const float*)d_in[4];
  const float* b_out  = (const float*)d_in[5];
  const float* ln1_g  = (const float*)d_in[6];
  const float* ln1_b  = (const float*)d_in[7];
  const float* ln2_g  = (const float*)d_in[8];
  const float* ln2_b  = (const float*)d_in[9];
  float* out = (float*)d_out;

  char* ws = (char*)d_ws;
  unsigned short* xn   = (unsigned short*)(ws);                      //  8 MB
  unsigned short* Wpt  = (unsigned short*)(ws + (size_t)( 8u << 20));//  8 MB
  unsigned short* Wot  = (unsigned short*)(ws + (size_t)(16u << 20));//  2 MB
  unsigned short* proj = (unsigned short*)(ws + (size_t)(18u << 20));// 32 MB
  unsigned short* Qh   = (unsigned short*)(ws + (size_t)(50u << 20));//  8 MB
  unsigned short* Kh   = (unsigned short*)(ws + (size_t)(58u << 20));//  8 MB
  unsigned short* Vt   = (unsigned short*)(ws + (size_t)(66u << 20));//  8 MB
  float*          x2   = (float*)        (ws + (size_t)(74u << 20)); // 16 MB
  unsigned short* xn2  = (unsigned short*)(ws + (size_t)(90u << 20));//  8 MB  (98 MB total)

  // weights -> bf16 B-operand layout
  k_wconv<<<(1024 * 4096 + 255) / 256, 256, 0, stream>>>(W_proj, Wpt, 1024, 4096);
  k_wconv<<<(1024 * 1024 + 255) / 256, 256, 0, stream>>>(W_out,  Wot, 1024, 1024);
  // LN1
  k_ln<<<BB * LL, 256, 0, stream>>>(x, ln1_g, ln1_b, xn);
  // proj = LN1(x) @ W_proj + b_proj   (bf16 out)
  {
    dim3 g(4096 / 16, 4096 / 128);
    k_gemm<<<g, 32, 0, stream>>>(xn, Wpt, b_proj, nullptr, proj, nullptr,
                                 4096, 4096, 1024, 0);
  }
  // split + RoPE + V transpose
  k_split_rope<<<(1 << 22) / 256, 256, 0, stream>>>(proj, Qh, Kh, Vt);
  // sigmoid attention fused with silu(U) gate and residual -> x2
  {
    dim3 g(LL / 16, NHH, BB);
    k_attn<<<g, 32, 0, stream>>>(Qh, Kh, Vt, proj, x, x2);
  }
  // LN2 and final GEMM with residual -> d_out
  k_ln<<<BB * LL, 256, 0, stream>>>(x2, ln2_g, ln2_b, xn2);
  {
    dim3 g(4096 / 16, 1024 / 128);
    k_gemm<<<g, 32, 0, stream>>>(xn2, Wot, b_out, x2, nullptr, out,
                                 4096, 1024, 1024, 1);
  }
}
